// AdaMemAttention_49194555408557
// MI455X (gfx1250) — matrix operand order repaired and checked by reading, not compiled, not run on hardware
//
#include <hip/hip_runtime.h>
#include <hip/hip_bf16.h>

typedef __bf16 bf16_t;
typedef __attribute__((ext_vector_type(16))) __bf16 v16bf;
typedef __attribute__((ext_vector_type(8)))  __bf16 v8bf;
typedef __attribute__((ext_vector_type(8)))  float  v8f;
typedef int i32x4 __attribute__((vector_size(16)));   // matches builtin's V4i

#define B_   4
#define N_   1568
#define C_   768
#define H_   12
#define D_   64
#define NB_  2048
#define NP_  1568
#define KB_  153      // int(512*0.3)
#define KP_  358      // int(512*0.7)
#define MTOT 2079     // KB_+KP_+N_
#define CHK  64       // staged key chunk
#define MPAD 2112     // 33*64, multiple of CHK
#define NCH  (MPAD / CHK)

#if defined(__gfx1250__) && __has_builtin(__builtin_amdgcn_global_load_async_to_lds_b128)
#define USE_ASYNC_LDS 1
#endif

#define AS1 __attribute__((address_space(1)))
#define AS3 __attribute__((address_space(3)))

__device__ __forceinline__ v8f wmma_bf16(v16bf a, v16bf b, v8f c) {
  return __builtin_amdgcn_wmma_f32_16x16x32_bf16(false, a, false, b, (short)0, c,
                                                 false, false);
}

// Two contiguous 8-element chunks -> 16-bit A-fragment lane layout.
__device__ __forceinline__ v16bf pack2(const bf16_t* p0, const bf16_t* p1) {
  v8bf lo = *(const v8bf*)p0;
  v8bf hi = *(const v8bf*)p1;
  v16bf r;
#pragma unroll
  for (int e = 0; e < 8; ++e) { r[e] = lo[e]; r[e + 8] = hi[e]; }
  return r;
}

__device__ __forceinline__ void wait_async(int pending8) {
#ifdef USE_ASYNC_LDS
#if __has_builtin(__builtin_amdgcn_s_wait_asynccnt)
  if (pending8) __builtin_amdgcn_s_wait_asynccnt(8);
  else          __builtin_amdgcn_s_wait_asynccnt(0);
#else
  if (pending8) asm volatile("s_wait_asynccnt 8" ::: "memory");
  else          asm volatile("s_wait_asynccnt 0" ::: "memory");
#endif
#else
  (void)pending8;
#endif
}

// Stage one 64-key chunk: K (8KB contiguous) + V^T (64 rows x 128B).
// 128 threads x 4 x 16B per tensor; async -> 8 ASYNCcnt ticks per wave.
__device__ __forceinline__ void stage_chunk(const bf16_t* __restrict__ kg,
                                            const bf16_t* __restrict__ vg,
                                            bf16_t* ks, bf16_t* vs, int tid) {
#ifdef USE_ASYNC_LDS
#pragma unroll
  for (int r = 0; r < 4; ++r) {
    int idx = r * 128 + tid;
    __builtin_amdgcn_global_load_async_to_lds_b128(
        (AS1 i32x4*)(void*)(kg + idx * 8),
        (AS3 i32x4*)(void*)(ks + idx * 8), 0, 0);
  }
#pragma unroll
  for (int r = 0; r < 4; ++r) {
    int idx = r * 128 + tid;
    int dd = idx >> 3, seg = idx & 7;
    __builtin_amdgcn_global_load_async_to_lds_b128(
        (AS1 i32x4*)(void*)(vg + (size_t)dd * MPAD + seg * 8),
        (AS3 i32x4*)(void*)(vs + dd * CHK + seg * 8), 0, 0);
  }
#else
#pragma unroll
  for (int r = 0; r < 4; ++r) {
    int idx = r * 128 + tid;
    *(v8bf*)(ks + idx * 8) = *(const v8bf*)(kg + idx * 8);
    int dd = idx >> 3, seg = idx & 7;
    *(v8bf*)(vs + dd * CHK + seg * 8) =
        *(const v8bf*)(vg + (size_t)dd * MPAD + seg * 8);
  }
#endif
}

// ---------------------------------------------------------------- fp32->bf16
__global__ void cvt_f32_to_bf16(const float* __restrict__ in,
                                bf16_t* __restrict__ out, int n) {
  int i = blockIdx.x * blockDim.x + threadIdx.x;
  if (i < n) out[i] = (bf16_t)in[i];
}

// ------------------------------------------------------------- QKV WMMA GEMM
__global__ void qkv_wmma_kernel(const bf16_t* __restrict__ xb,
                                const bf16_t* __restrict__ wb,
                                bf16_t* __restrict__ qb,
                                bf16_t* __restrict__ kcat,
                                bf16_t* __restrict__ vcatT,
                                float*  __restrict__ q1) {
  const int lane = threadIdx.x & 31;
  const int wid  = threadIdx.x >> 5;
  const int gw   = blockIdx.x * 8 + wid;
  const int NT   = (3 * C_) / 16;          // 144 col tiles
  const int mt   = gw / NT;
  const int nt   = gw % NT;
  if (mt >= (B_ * N_) / 16) return;
  const int half = lane >> 4;
  const int l16  = lane & 15;
  const bf16_t* arow = xb + (size_t)(mt * 16 + l16) * C_;
  const bf16_t* brow = wb + (size_t)(nt * 16 + l16) * C_;
  v8f acc = {};
#pragma unroll 4
  for (int kb = 0; kb < C_; kb += 32) {
    __builtin_prefetch(arow + kb + 128, 0, 3);
    __builtin_prefetch(brow + kb + 128, 0, 3);
    v16bf a = pack2(arow + kb + half * 16, arow + kb + half * 16 + 8);
    v16bf b = *(const v16bf*)(brow + kb + half * 16);
    acc = wmma_bf16(a, b, acc);
  }
  const int col = nt * 16 + l16;
  const int s   = col / C_;                 // 0=q 1=k 2=v
  const int rem = col % C_;
  const int h   = rem / D_;
  const int dd  = rem % D_;
#pragma unroll
  for (int i = 0; i < 8; ++i) {
    const int m  = mt * 16 + i + half * 8;  // C-layout row
    const int b_ = m / N_;
    const int nn = m % N_;
    const int bh = b_ * H_ + h;
    float v = acc[i];
    if (s == 0) {
      qb[((size_t)bh * N_ + nn) * D_ + dd] = (bf16_t)v;
      if (nn == 0) q1[bh * D_ + dd] = v;    // fp32 q1 for selection scores
    } else if (s == 1) {
      kcat[((size_t)bh * MPAD + (KB_ + KP_) + nn) * D_ + dd] = (bf16_t)v;
    } else {
      vcatT[((size_t)bh * D_ + dd) * MPAD + (KB_ + KP_) + nn] = (bf16_t)v;
    }
  }
}

// ----------------------------------------------------- selection scores (f32)
__global__ void score_kernel(const float* __restrict__ q1,
                             const float* __restrict__ ck,
                             float* __restrict__ sc, int srcN) {
  const int bh = blockIdx.x;
  __shared__ float qs[D_];
  if (threadIdx.x < D_) qs[threadIdx.x] = q1[bh * D_ + threadIdx.x];
  __syncthreads();
  for (int j = threadIdx.x; j < NB_; j += blockDim.x) {
    float s;
    if (j < srcN) {
      const float* kr = ck + ((size_t)bh * srcN + j) * D_;
      s = 0.f;
#pragma unroll 8
      for (int d = 0; d < D_; ++d) s += qs[d] * kr[d];
    } else {
      s = -1e30f;
    }
    sc[(size_t)bh * NB_ + j] = s;
  }
}

// ------------------------------------------- top-k (bitonic sort) + KV gather
__global__ void topk_gather_kernel(const float* __restrict__ sc,
                                   const float* __restrict__ srck,
                                   const float* __restrict__ srcv,
                                   bf16_t* __restrict__ kcat,
                                   bf16_t* __restrict__ vcatT,
                                   int srcN, int kk, int dstOff, int doPad) {
  const int bh = blockIdx.x;
  __shared__ float sv[NB_];
  __shared__ int   si[NB_];
  for (int t = threadIdx.x; t < NB_; t += blockDim.x) {
    sv[t] = sc[(size_t)bh * NB_ + t];
    si[t] = t;
  }
  __syncthreads();
  for (int k = 2; k <= NB_; k <<= 1) {        // bitonic sort, descending
    for (int j = k >> 1; j > 0; j >>= 1) {
      for (int t = threadIdx.x; t < NB_; t += blockDim.x) {
        int ixj = t ^ j;
        if (ixj > t) {
          bool up = ((t & k) != 0);
          float v0 = sv[t], v1 = sv[ixj];
          if ((v0 > v1) == up) {
            int i0 = si[t];
            sv[t] = v1; sv[ixj] = v0;
            si[t] = si[ixj]; si[ixj] = i0;
          }
        }
      }
      __syncthreads();
    }
  }
  for (int t = threadIdx.x; t < kk * D_; t += blockDim.x) {
    int r = t / D_, dd = t % D_;
    int row = si[r];
    float kv = srck[((size_t)bh * srcN + row) * D_ + dd];
    float vv = srcv[((size_t)bh * srcN + row) * D_ + dd];
    kcat[((size_t)bh * MPAD + dstOff + r) * D_ + dd]  = (bf16_t)kv;
    vcatT[((size_t)bh * D_ + dd) * MPAD + dstOff + r] = (bf16_t)vv;
  }
  if (doPad) {                                 // zero rows [MTOT, MPAD)
    for (int t = threadIdx.x; t < (MPAD - MTOT) * D_; t += blockDim.x) {
      int r = t / D_, dd = t % D_;
      kcat[((size_t)bh * MPAD + MTOT + r) * D_ + dd]  = (bf16_t)0.f;
      vcatT[((size_t)bh * D_ + dd) * MPAD + MTOT + r] = (bf16_t)0.f;
    }
  }
}

// ------------------------- flash attention: async-LDS staged K/V + WMMA bf16
__global__ void flash_attn_kernel(const bf16_t* __restrict__ qb,
                                  const bf16_t* __restrict__ kcat,
                                  const bf16_t* __restrict__ vcatT,
                                  bf16_t* __restrict__ attn) {
  const int bh = blockIdx.x;
  const int b_ = bh / H_;
  const int h  = bh % H_;
  const int tid  = threadIdx.x;
  const int lane = tid & 31;
  const int wid  = tid >> 5;
  const int qt_raw = blockIdx.y * 4 + wid;    // 16-row query tile
  const bool active = qt_raw < (N_ / 16);
  const int qt = active ? qt_raw : (N_ / 16 - 1);
  const int half = lane >> 4;
  const int l16  = lane & 15;

  __shared__ __align__(32) bf16_t ktile[2][CHK * D_];   // [m][d]
  __shared__ __align__(32) bf16_t vtile[2][D_ * CHK];   // [d][m]
  __shared__ __align__(32) bf16_t pst[4][16 * 32];

  const bf16_t* qrow = qb + ((size_t)bh * N_ + qt * 16 + l16) * D_;
  v16bf qa[2];
#pragma unroll
  for (int ks = 0; ks < 2; ++ks)
    qa[ks] = pack2(qrow + ks * 32 + half * 16, qrow + ks * 32 + half * 16 + 8);

  v8f o0 = {}, o1 = {}, o2 = {}, o3 = {};
  float rm[8], rl[8];
#pragma unroll
  for (int i = 0; i < 8; ++i) { rm[i] = -1e30f; rl[i] = 0.f; }
  const float scale = 0.125f;                 // 64^-0.5
  bf16_t* myp = pst[wid];
  const bf16_t* kg0 = kcat + (size_t)bh * MPAD * D_;
  const bf16_t* vg0 = vcatT + (size_t)bh * D_ * MPAD;

  stage_chunk(kg0, vg0, ktile[0], vtile[0], tid);       // prologue: chunk 0

  for (int ch = 0; ch < NCH; ++ch) {
    const int cur = ch & 1;
    if (ch + 1 < NCH)
      stage_chunk(kg0 + (size_t)(ch + 1) * CHK * D_, vg0 + (ch + 1) * CHK,
                  ktile[cur ^ 1], vtile[cur ^ 1], tid);
    wait_async(ch + 1 < NCH);                 // drain current chunk's copies
    __syncthreads();
    const bf16_t* kbuf = ktile[cur];
    const bf16_t* vbuf = vtile[cur];
    const int mb = ch * CHK;
#pragma unroll
    for (int sub = 0; sub < 2; ++sub) {
      const int ml = sub * 32;
      v8f s0 = {}, s1 = {};
#pragma unroll
      for (int ks = 0; ks < 2; ++ks) {
        const bf16_t* k0 = kbuf + (ml + l16) * D_ + ks * 32 + half * 16;
        const bf16_t* k1 = kbuf + (ml + 16 + l16) * D_ + ks * 32 + half * 16;
        s0 = wmma_bf16(qa[ks], *(const v16bf*)k0, s0);
        s1 = wmma_bf16(qa[ks], *(const v16bf*)k1, s1);
      }
      float p0[8], p1[8], alpha[8];
#pragma unroll
      for (int i = 0; i < 8; ++i) {
        float a = s0[i] * scale;
        float c = s1[i] * scale;
        if (mb + ml + l16 >= MTOT)      a = -1e30f;    // padding mask
        if (mb + ml + 16 + l16 >= MTOT) c = -1e30f;
        float cm = fmaxf(a, c);
#pragma unroll
        for (int d = 1; d < 16; d <<= 1) cm = fmaxf(cm, __shfl_xor(cm, d, 32));
        float rmn = fmaxf(rm[i], cm);
        alpha[i] = __expf(rm[i] - rmn);
        rm[i] = rmn;
        float e0 = __expf(a - rmn);
        float e1 = __expf(c - rmn);
        float cs = e0 + e1;
#pragma unroll
        for (int d = 1; d < 16; d <<= 1) cs += __shfl_xor(cs, d, 32);
        rl[i] = rl[i] * alpha[i] + cs;
        p0[i] = e0; p1[i] = e1;
      }
#pragma unroll
      for (int i = 0; i < 8; ++i) {
        o0[i] *= alpha[i]; o1[i] *= alpha[i];
        o2[i] *= alpha[i]; o3[i] *= alpha[i];
        int row = i + half * 8;               // C-layout row -> LDS P tile
        myp[row * 32 + l16]      = (bf16_t)p0[i];
        myp[row * 32 + 16 + l16] = (bf16_t)p1[i];
      }
      asm volatile("s_wait_dscnt 0" ::: "memory");  // DS RAW, same wave
      v16bf pa = pack2(myp + l16 * 32 + half * 16,
                       myp + l16 * 32 + half * 16 + 8);
      const bf16_t* vb = vbuf + l16 * CHK + ml + half * 16;
      o0 = wmma_bf16(pa, *(const v16bf*)(vb + 0 * 16 * CHK), o0);
      o1 = wmma_bf16(pa, *(const v16bf*)(vb + 1 * 16 * CHK), o1);
      o2 = wmma_bf16(pa, *(const v16bf*)(vb + 2 * 16 * CHK), o2);
      o3 = wmma_bf16(pa, *(const v16bf*)(vb + 3 * 16 * CHK), o3);
    }
    __syncthreads();                          // buffer reuse fence
  }

  if (active) {
    const int nq0 = qt * 16;
#pragma unroll
    for (int i = 0; i < 8; ++i) {
      int row = i + half * 8;
      float inv = 1.0f / rl[i];
      size_t base = ((size_t)(b_ * N_ + nq0 + row)) * C_ + h * D_ + l16;
      attn[base +  0] = (bf16_t)(o0[i] * inv);
      attn[base + 16] = (bf16_t)(o1[i] * inv);
      attn[base + 32] = (bf16_t)(o2[i] * inv);
      attn[base + 48] = (bf16_t)(o3[i] * inv);
    }
  }
}

// ------------------------------------------------------- output proj (WMMA)
__global__ void proj_wmma_kernel(const bf16_t* __restrict__ ab,
                                 const bf16_t* __restrict__ wb,
                                 const float* __restrict__ bias,
                                 float* __restrict__ out) {
  const int lane = threadIdx.x & 31;
  const int wid  = threadIdx.x >> 5;
  const int gw   = blockIdx.x * 8 + wid;
  const int NT   = C_ / 16;                // 48 col tiles
  const int mt   = gw / NT;
  const int nt   = gw % NT;
  if (mt >= (B_ * N_) / 16) return;
  const int half = lane >> 4;
  const int l16  = lane & 15;
  const bf16_t* arow = ab + (size_t)(mt * 16 + l16) * C_;
  const bf16_t* brow = wb + (size_t)(nt * 16 + l16) * C_;
  v8f acc = {};
#pragma unroll 4
  for (int kb = 0; kb < C_; kb += 32) {
    __builtin_prefetch(arow + kb + 128, 0, 3);
    __builtin_prefetch(brow + kb + 128, 0, 3);
    v16bf a = pack2(arow + kb + half * 16, arow + kb + half * 16 + 8);
    v16bf b = *(const v16bf*)(brow + kb + half * 16);
    acc = wmma_bf16(a, b, acc);
  }
  const int col = nt * 16 + l16;
  const float bv = bias[col];
#pragma unroll
  for (int i = 0; i < 8; ++i) {
    int m = mt * 16 + i + half * 8;
    out[(size_t)m * C_ + col] = acc[i] + bv;
  }
}

// --------------------------------------------------------------------------
extern "C" void kernel_launch(void* const* d_in, const int* in_sizes, int n_in,
                              void* d_out, int out_size, void* d_ws,
                              size_t ws_size, hipStream_t stream) {
  const float* x      = (const float*)d_in[0];
  const float* bank_k = (const float*)d_in[1];
  const float* bank_v = (const float*)d_in[2];
  const float* prev_k = (const float*)d_in[3];
  const float* prev_v = (const float*)d_in[4];
  const float* w_qkv  = (const float*)d_in[5];
  const float* w_proj = (const float*)d_in[6];
  const float* b_proj = (const float*)d_in[7];
  float* out = (float*)d_out;

  char* ws = (char*)d_ws;
  size_t off = 0;
  auto alloc = [&](size_t bytes) -> char* {
    char* p = ws + off;
    off = (off + bytes + 255) & ~(size_t)255;
    return p;
  };
  const size_t BH = (size_t)B_ * H_;
  bf16_t* xb    = (bf16_t*)alloc((size_t)B_ * N_ * C_ * 2);
  bf16_t* wqkvb = (bf16_t*)alloc((size_t)3 * C_ * C_ * 2);
  bf16_t* wprjb = (bf16_t*)alloc((size_t)C_ * C_ * 2);
  bf16_t* qb    = (bf16_t*)alloc(BH * N_ * D_ * 2);
  float*  q1    = (float*) alloc(BH * D_ * 4);
  bf16_t* kcat  = (bf16_t*)alloc(BH * MPAD * D_ * 2);
  bf16_t* vcatT = (bf16_t*)alloc(BH * D_ * MPAD * 2);
  float*  scb   = (float*) alloc(BH * NB_ * 4);
  float*  scp   = (float*) alloc(BH * NB_ * 4);
  bf16_t* attnb = (bf16_t*)alloc((size_t)B_ * N_ * C_ * 2);

  {  // 1) fp32 -> bf16 operand conversion
    int n1 = B_ * N_ * C_, n2 = 3 * C_ * C_, n3 = C_ * C_;
    cvt_f32_to_bf16<<<(n1 + 255) / 256, 256, 0, stream>>>(x, xb, n1);
    cvt_f32_to_bf16<<<(n2 + 255) / 256, 256, 0, stream>>>(w_qkv, wqkvb, n2);
    cvt_f32_to_bf16<<<(n3 + 255) / 256, 256, 0, stream>>>(w_proj, wprjb, n3);
  }
  {  // 2) QKV GEMM (WMMA bf16): 392 x 144 tiles, 8 waves/block
    int waves = ((B_ * N_) / 16) * ((3 * C_) / 16);
    qkv_wmma_kernel<<<(waves + 7) / 8, 256, 0, stream>>>(xb, wqkvb, qb, kcat,
                                                         vcatT, q1);
  }
  // 3) selection scores (fp32)
  score_kernel<<<B_ * H_, 256, 0, stream>>>(q1, bank_k, scb, NB_);
  score_kernel<<<B_ * H_, 256, 0, stream>>>(q1, prev_k, scp, NP_);
  // 4) top-k + gather into Kcat/VcatT
  topk_gather_kernel<<<B_ * H_, 256, 0, stream>>>(scb, bank_k, bank_v, kcat,
                                                  vcatT, NB_, KB_, 0, 1);
  topk_gather_kernel<<<B_ * H_, 256, 0, stream>>>(scp, prev_k, prev_v, kcat,
                                                  vcatT, NP_, KP_, KB_, 0);
  {  // 5) flash attention (async-LDS staged, WMMA bf16, fp32 softmax)
    dim3 grid(B_ * H_, (N_ / 16 + 3) / 4);
    flash_attn_kernel<<<grid, 128, 0, stream>>>(qb, kcat, vcatT, attnb);
  }
  {  // 6) output projection (WMMA bf16) + bias
    int waves = ((B_ * N_) / 16) * (C_ / 16);
    proj_wmma_kernel<<<(waves + 7) / 8, 256, 0, stream>>>(attnb, wprjb, b_proj,
                                                          out);
  }
}
